// Net_n_mnist_65163243815586
// MI455X (gfx1250) — matrix operand order repaired and checked by reading, not compiled, not run on hardware
//
#include <hip/hip_runtime.h>
#include <hip/hip_bf16.h>

typedef __attribute__((ext_vector_type(16))) _Float16 v16h;
typedef __attribute__((ext_vector_type(8)))  float    v8f;

#define HASH_BITS 21
#define HASH_ENTRIES (1u << HASH_BITS)

// ---------------------------------------------------------------- utilities

__device__ __forceinline__ void atomicMaxF(float* addr, float val) {
    // monotonic-in-float-order trick: non-negative -> int max, negative -> uint min
    if (val >= 0.0f) atomicMax((int*)addr, __float_as_int(val));
    else             atomicMin((unsigned int*)addr, __float_as_uint(val));
}

__device__ __forceinline__ float eluf(float t) { return t > 0.0f ? t : expm1f(t); }

__global__ void k_fill(float* p, float v, long n) {
    for (long i = blockIdx.x * (long)blockDim.x + threadIdx.x; i < n;
         i += (long)gridDim.x * blockDim.x) p[i] = v;
}

// ------------------------------------------------- weight packing for WMMA B
// Wp[((k*KC + kc)*cout + c)*32 + h*16 + j] = W[k, kc*32 + 16*(j/8) + h*8 + j%8, c]
// -> each lane's B fragment (16 f16 for its column/half) is one contiguous,
//    32B-aligned v16h load, and the fragment index is linear in (k*KC + kc).
__global__ void k_pack_w(const float* __restrict__ W, _Float16* __restrict__ Wp,
                         int cin, int cout, int Ktot) {
    const int KC = cin >> 5;
    const long total = (long)Ktot * cin * cout;
    for (long i = blockIdx.x * (long)blockDim.x + threadIdx.x; i < total;
         i += (long)gridDim.x * blockDim.x) {
        int  j  = (int)(i & 15);
        int  h  = (int)((i >> 4) & 1);
        long t  = i >> 5;                 // (k*KC + kc)*cout + c
        int  c  = (int)(t % cout);
        long u  = t / cout;               // k*KC + kc
        int  kc = (int)(u % KC);
        int  k  = (int)(u / KC);
        int  kk = kc * 32 + (j >> 3) * 16 + h * 8 + (j & 7);
        Wp[i] = (_Float16)W[((size_t)k * cin + kk) * cout + c];
    }
}

// ------------------------------------------------- WMMA transforms
// CONTRACT: cin % 32 == 0, cout % 16 == 0, n % 16 == 0 (all call sites comply:
// n in {65536, 18496, 5184, 64}) -> no predication anywhere, straight-line
// loads and stores.

__device__ __forceinline__ void load_a_frag(const float* __restrict__ p, v16h& a) {
    // p points at xrow + kb + half*8 ; two contiguous 8-float runs
    #pragma unroll
    for (int j = 0; j < 8; ++j) {
        a[j]     = (_Float16)p[j];
        a[8 + j] = (_Float16)p[16 + j];
    }
}

// Multi-kernel-point transform: A fragments held in registers across all Ktot.
// Y[n, k, cout] = X[n,:] @ W[k]  (CHUNKS = cin/32, cin in {32,64,128})
template<int CHUNKS, bool OUT_HALF>
__global__ void k_transform_multik(const float* __restrict__ X, const _Float16* __restrict__ Wp,
                                   void* __restrict__ Y, int n, int cout, int Ktot) {
    const int cin = CHUNKS * 32;
    const int tm = blockIdx.x, tn = blockIdx.y;
    const int lane = threadIdx.x;
    const int half = lane >> 4;
    const int lc = lane & 15;
    const int row = tm * 16 + lc;            // n % 16 == 0 -> always in range
    const int col = tn * 16 + lc;

    v16h A[CHUNKS];
    {
        const float* xrow = X + (size_t)row * cin;
        #pragma unroll
        for (int c0 = 0; c0 < CHUNKS; ++c0)
            load_a_frag(xrow + c0 * 32 + half * 8, A[c0]);
    }
    const int m0 = tm * 16 + half * 8;
    // B fragment offset is linear in (k*CHUNKS + c0): step = cout*32 halves
    size_t boff = (((size_t)col) << 5) + (size_t)half * 16;
    const size_t bstep = ((size_t)cout) << 5;
    // output offsets: elem(r, k) = obase + k*cout + r*(Ktot*cout)
    size_t obase = ((size_t)m0 * Ktot) * cout + col;
    const size_t orstride = (size_t)Ktot * cout;

    for (int k = 0; k < Ktot; ++k) {
        v8f acc = {};
        #pragma unroll
        for (int c0 = 0; c0 < CHUNKS; ++c0) {
            const v16h b = *(const v16h*)(Wp + boff);
            boff += bstep;
            acc = __builtin_amdgcn_wmma_f32_16x16x32_f16(false, A[c0], false, b,
                                                         (short)0, acc, false, false);
        }
        #pragma unroll
        for (int r = 0; r < 8; ++r) {
            size_t o = obase + (size_t)r * orstride;
            if (OUT_HALF) ((_Float16*)Y)[o] = (_Float16)acc[r];
            else          ((float*)Y)[o]    = acc[r];
        }
        obase += cout;
    }
}

// Streaming transform for large cin (FC1: cin=3200, Ktot=1).
template<bool OUT_HALF>
__global__ void k_transform_stream(const float* __restrict__ X, const _Float16* __restrict__ Wp,
                                   void* __restrict__ Y, int n, int cin, int cout) {
    const int tm = blockIdx.x, tn = blockIdx.y;
    const int lane = threadIdx.x;
    const int half = lane >> 4;
    const int lc = lane & 15;
    const int row = tm * 16 + lc;
    const int col = tn * 16 + lc;
    const float* xrow = X + (size_t)row * cin;
    const int KC = cin >> 5;
    size_t boff = (((size_t)col) << 5) + (size_t)half * 16;
    const size_t bstep = ((size_t)cout) << 5;
    v8f acc = {};
    for (int kc = 0; kc < KC; ++kc) {
        v16h a, b;
        load_a_frag(xrow + kc * 32 + half * 8, a);
        b = *(const v16h*)(Wp + boff);
        boff += bstep;
        acc = __builtin_amdgcn_wmma_f32_16x16x32_f16(false, a, false, b,
                                                     (short)0, acc, false, false);
    }
    const int m0 = tm * 16 + half * 8;
    size_t obase = (size_t)m0 * cout + col;
    #pragma unroll
    for (int r = 0; r < 8; ++r) {
        size_t o = obase + (size_t)r * cout;
        if (OUT_HALF) ((_Float16*)Y)[o] = (_Float16)acc[r];
        else          ((float*)Y)[o]    = acc[r];
    }
}

// cin == 1 special case (first layer): y = x[n] * W[k, 0, c]
template<bool OUT_HALF>
__global__ void k_transform_cin1(const float* __restrict__ X, const float* __restrict__ W,
                                 void* __restrict__ Y, int n, int cout, int Ktot) {
    int i = blockIdx.x * blockDim.x + threadIdx.x;
    if (i >= n * cout) return;
    int row = i / cout, c = i % cout;
    float x = X[row];
    for (int k = 0; k < Ktot; ++k) {
        float v = x * W[k * cout + c];
        size_t o = ((size_t)row * Ktot + k) * cout + c;
        if (OUT_HALF) ((_Float16*)Y)[o] = (_Float16)v;
        else          ((float*)Y)[o]    = v;
    }
}

// ------------------------------------------------- fused message + segment_sum

__global__ void k_edge_cnt(const int* __restrict__ dst, const float* __restrict__ ev,
                           float* __restrict__ cnt, int E) {
    int e = blockIdx.x * blockDim.x + threadIdx.x;
    if (e >= E) return;
    float w = ev ? ev[e] : 1.0f;
    if (w > 0.0f) atomicAdd(cnt + dst[e], w);
}

__global__ void k_edge_msg(const _Float16* __restrict__ Y, const float* __restrict__ pseudo,
                           const int* __restrict__ src, const int* __restrict__ dst,
                           const float* __restrict__ ev, float* __restrict__ agg,
                           int E, int cout, int Ktot) {
    const int cg = cout >> 2;                 // 4 channels per thread
    const long total = (long)E * cg;
    for (long i = blockIdx.x * (long)blockDim.x + threadIdx.x; i < total;
         i += (long)gridDim.x * blockDim.x) {
        int e = (int)(i / cg);
        int c = (int)(i % cg) * 4;
        float w = ev ? ev[e] : 1.0f;
        if (w <= 0.0f) continue;
        int s = src[e], d = dst[e];
        float m0 = 0.f, m1 = 0.f, m2 = 0.f, m3 = 0.f;
        if (Ktot == 1) {
            const _Float16* yp = Y + (size_t)s * cout + c;
            m0 = (float)yp[0]; m1 = (float)yp[1]; m2 = (float)yp[2]; m3 = (float)yp[3];
        } else {
            // degree-1 open B-spline basis over [0,1]^2, K=5 per dim
            float vx = pseudo[2 * e] * 4.0f, vy = pseudo[2 * e + 1] * 4.0f;
            int ix = min(max((int)floorf(vx), 0), 3);
            int iy = min(max((int)floorf(vy), 0), 3);
            float fx = vx - (float)ix, fy = vy - (float)iy;
            float bs[4] = { (1.f - fx) * (1.f - fy), fx * (1.f - fy),
                            (1.f - fx) * fy,          fx * fy };
            int id[4] = { ix + 5 * iy, ix + 1 + 5 * iy, ix + 5 * (iy + 1), ix + 1 + 5 * (iy + 1) };
            const _Float16* yb = Y + ((size_t)s * Ktot) * cout + c;
            #pragma unroll
            for (int t = 0; t < 4; ++t) {
                const _Float16* yp = yb + (size_t)id[t] * cout;
                float bt = bs[t];
                m0 += bt * (float)yp[0]; m1 += bt * (float)yp[1];
                m2 += bt * (float)yp[2]; m3 += bt * (float)yp[3];
            }
        }
        float* ap = agg + (size_t)d * cout + c;
        atomicAdd(ap + 0, m0 * w); atomicAdd(ap + 1, m1 * w);
        atomicAdd(ap + 2, m2 * w); atomicAdd(ap + 3, m3 * w);
    }
}

// mean-normalize + root-transform + bias + eval BN (+ optional ELU)
__global__ void k_epilogue(const float* __restrict__ agg, const float* __restrict__ cnt,
                           const float* __restrict__ xr, const float* __restrict__ bias,
                           const float* __restrict__ g, const float* __restrict__ bb,
                           const float* __restrict__ m, const float* __restrict__ v,
                           float* __restrict__ out, int n, int C, int do_elu) {
    int i = blockIdx.x * blockDim.x + threadIdx.x;
    if (i >= n * C) return;
    int row = i / C, c = i % C;
    float t = agg[i] / fmaxf(cnt[row], 1.0f) + xr[i] + bias[c];
    t = (t - m[c]) * rsqrtf(v[c] + 1e-5f) * g[c] + bb[c];
    if (do_elu) t = eluf(t);
    out[i] = t;
}

__global__ void k_add_elu(const float* a, const float* b, float* out, int n) {
    int i = blockIdx.x * blockDim.x + threadIdx.x;
    if (i < n) out[i] = eluf(a[i] + b[i]);
}

// ------------------------------------------------- voxel pooling

__global__ void k_pool_scatter(const float* __restrict__ x, const float* __restrict__ pos,
                               const float* __restrict__ nvalid,
                               const int* __restrict__ batch_arr, int batch_div,
                               float size, int nb, int C,
                               float* __restrict__ xc, float* __restrict__ posc,
                               float* __restrict__ cnt, int* __restrict__ cl_out, int n) {
    int i = blockIdx.x * blockDim.x + threadIdx.x;
    if (i >= n) return;
    float nv = nvalid ? nvalid[i] : 1.0f;
    float px = pos[2 * i], py = pos[2 * i + 1];
    int b = batch_arr ? batch_arr[i] : (i / batch_div);
    int gx = min(max((int)floorf(px / size), 0), nb - 1);
    int gy = min(max((int)floorf(py / size), 0), nb - 1);
    int cl = b * nb * nb + gx * nb + gy;
    cl_out[i] = cl;
    if (nv <= 0.0f) return;
    atomicAdd(cnt + cl, nv);
    if (posc) { atomicAdd(posc + 2 * cl, px * nv); atomicAdd(posc + 2 * cl + 1, py * nv); }
    const float* xrow = x + (size_t)i * C;
    float* xcr = xc + (size_t)cl * C;
    for (int c = 0; c < C; ++c) atomicMaxF(xcr + c, xrow[c]);
}

__global__ void k_pool_finish(float* __restrict__ xc, float* __restrict__ posc,
                              const float* __restrict__ cnt, float* __restrict__ cvalid,
                              int C, int nc) {
    int i = blockIdx.x * blockDim.x + threadIdx.x;
    if (i >= nc) return;
    float ct = cnt[i];
    if (posc) { float inv = 1.0f / fmaxf(ct, 1.0f); posc[2 * i] *= inv; posc[2 * i + 1] *= inv; }
    if (cvalid) cvalid[i] = ct > 0.0f ? 1.0f : 0.0f;
    float* xr = xc + (size_t)i * C;
    for (int c = 0; c < C; ++c) { float v = xr[c]; xr[c] = (v < -1e37f) ? 0.0f : v; }
}

// edge remap + self-loop drop + hash-table dedup + cart + global |cart| max
__global__ void k_edge_remap(const int* __restrict__ src, const int* __restrict__ dst,
                             const float* __restrict__ ev, const int* __restrict__ cl,
                             const float* __restrict__ posc, int nc,
                             int* __restrict__ nsrc, int* __restrict__ ndst,
                             float* __restrict__ evout, float* __restrict__ cart,
                             unsigned long long* __restrict__ hash, float* __restrict__ mx,
                             int E) {
    int e = blockIdx.x * blockDim.x + threadIdx.x;
    if (e >= E) return;
    int s = cl[src[e]], d = cl[dst[e]];
    nsrc[e] = s; ndst[e] = d;
    float w = ev ? ev[e] : 1.0f;
    if (s == d) w = 0.0f;
    if (w > 0.0f) {
        unsigned long long key = (unsigned long long)s * (unsigned)nc + (unsigned)d + 1ull;
        unsigned h = (unsigned)((key * 0x9E3779B97F4A7C15ull) >> (64 - HASH_BITS));
        const unsigned mask = HASH_ENTRIES - 1u;
        for (;;) {
            unsigned long long prev = atomicCAS(hash + h, 0ull, key);
            if (prev == 0ull) break;                 // first occurrence: keep
            if (prev == key) { w = 0.0f; break; }    // duplicate: mask out
            h = (h + 1u) & mask;
        }
    }
    evout[e] = w;
    float cx = posc[2 * s] - posc[2 * d];
    float cy = posc[2 * s + 1] - posc[2 * d + 1];
    cart[2 * e] = cx; cart[2 * e + 1] = cy;
    if (w > 0.0f) { atomicMaxF(mx, fabsf(cx)); atomicMaxF(mx, fabsf(cy)); }
}

__global__ void k_pseudo_norm(float* __restrict__ ps, const float* __restrict__ mx, int E) {
    int e = blockIdx.x * blockDim.x + threadIdx.x;
    if (e >= E) return;
    float inv = 0.5f / fmaxf(mx[0], 1e-12f);
    ps[2 * e]     = ps[2 * e] * inv + 0.5f;
    ps[2 * e + 1] = ps[2 * e + 1] * inv + 0.5f;
}

// ------------------------------------------------- classifier head

__global__ void k_fc1_ep(const float* __restrict__ fco, const float* __restrict__ bias,
                         const float* __restrict__ g, const float* __restrict__ bb,
                         const float* __restrict__ m, const float* __restrict__ v,
                         float* __restrict__ out, int n) {
    int i = blockIdx.x * blockDim.x + threadIdx.x;
    if (i >= n) return;
    int c = i & 127;
    float t = eluf(fco[i] + bias[c]);
    out[i] = (t - m[c]) * rsqrtf(v[c] + 1e-5f) * g[c] + bb[c];
}

__global__ void k_head(const float* __restrict__ h, const float* __restrict__ W,
                       const float* __restrict__ b, float* __restrict__ out) {
    int r = threadIdx.x;
    if (r >= 64) return;
    float lo[10];
    #pragma unroll
    for (int j = 0; j < 10; ++j) lo[j] = b[j];
    const float* hr = h + r * 128;
    for (int k = 0; k < 128; ++k) {
        float hv = hr[k];
        const float* wr = W + k * 10;
        #pragma unroll
        for (int j = 0; j < 10; ++j) lo[j] += hv * wr[j];
    }
    float mx = lo[0];
    #pragma unroll
    for (int j = 1; j < 10; ++j) mx = fmaxf(mx, lo[j]);
    float ssum = 0.f;
    #pragma unroll
    for (int j = 0; j < 10; ++j) ssum += expf(lo[j] - mx);
    float ls = logf(ssum);
    #pragma unroll
    for (int j = 0; j < 10; ++j) out[r * 10 + j] = lo[j] - mx - ls;
}

// ---------------------------------------------------------------- host side

struct BN { const float *g, *b, *m, *v; };

static void launch_transform(const float* X, const float* W, void* Y, bool out_half,
                             int n, int cin, int cout, int Ktot, _Float16* wpack,
                             hipStream_t stream) {
    if (cin == 1) {
        int nt = n * cout;
        if (out_half) k_transform_cin1<true ><<<(nt + 255) / 256, 256, 0, stream>>>(X, W, Y, n, cout, Ktot);
        else          k_transform_cin1<false><<<(nt + 255) / 256, 256, 0, stream>>>(X, W, Y, n, cout, Ktot);
        return;
    }
    long wtot = (long)Ktot * cin * cout;
    long wb = (wtot + 255) / 256; int wblocks = (int)(wb < 2048 ? wb : 2048);
    k_pack_w<<<wblocks, 256, 0, stream>>>(W, wpack, cin, cout, Ktot);
    dim3 g((n + 15) / 16, cout / 16);
    if (cin == 32) {
        if (out_half) k_transform_multik<1, true ><<<g, 32, 0, stream>>>(X, wpack, Y, n, cout, Ktot);
        else          k_transform_multik<1, false><<<g, 32, 0, stream>>>(X, wpack, Y, n, cout, Ktot);
    } else if (cin == 64) {
        if (out_half) k_transform_multik<2, true ><<<g, 32, 0, stream>>>(X, wpack, Y, n, cout, Ktot);
        else          k_transform_multik<2, false><<<g, 32, 0, stream>>>(X, wpack, Y, n, cout, Ktot);
    } else if (cin == 128) {
        if (out_half) k_transform_multik<4, true ><<<g, 32, 0, stream>>>(X, wpack, Y, n, cout, Ktot);
        else          k_transform_multik<4, false><<<g, 32, 0, stream>>>(X, wpack, Y, n, cout, Ktot);
    } else { // large cin, Ktot == 1 (FC1)
        if (out_half) k_transform_stream<true ><<<g, 32, 0, stream>>>(X, wpack, Y, n, cin, cout);
        else          k_transform_stream<false><<<g, 32, 0, stream>>>(X, wpack, Y, n, cin, cout);
    }
}

static void conv_pass(const float* X, int n, int cin, int cout, int Ktot,
                      const float* W, const float* root, const float* bias, BN bn,
                      const float* pseudo, const int* src, const int* dst, const float* ev,
                      const float* cnt, _Float16* yh, _Float16* wpack,
                      float* xr, float* agg, float* out,
                      int do_elu, int E, hipStream_t stream) {
    launch_transform(X, W,    yh, true,  n, cin, cout, Ktot, wpack, stream);
    launch_transform(X, root, xr, false, n, cin, cout, 1,    wpack, stream);
    (void)hipMemsetAsync(agg, 0, (size_t)n * cout * sizeof(float), stream);
    long total = (long)E * (cout / 4);
    long want = (total + 255) / 256;
    int blocks = (int)(want < 8192 ? want : 8192);
    if (blocks < 1) blocks = 1;
    k_edge_msg<<<blocks, 256, 0, stream>>>(yh, pseudo, src, dst, ev, agg, E, cout, Ktot);
    int nC = n * cout;
    k_epilogue<<<(nC + 255) / 256, 256, 0, stream>>>(agg, cnt, xr, bias,
                                                     bn.g, bn.b, bn.m, bn.v, out, n, cout, do_elu);
}

static void run_res_block(const float* xin, int n, int cin, int cout,
                          const float* pseudo, const int* src, const int* dst, const float* ev,
                          const float* const* P, int base,
                          _Float16* yh, _Float16* wpack, float* xr, float* agg, float* cnt,
                          float* h1, float* h2, float* sb, float* out,
                          int E, hipStream_t stream) {
    (void)hipMemsetAsync(cnt, 0, (size_t)n * sizeof(float), stream);
    k_edge_cnt<<<(E + 255) / 256, 256, 0, stream>>>(dst, ev, cnt, E);
    BN bn1{ P[base + 3],  P[base + 2],  P[base + 4],  P[base + 5]  };
    BN bn2{ P[base + 7],  P[base + 6],  P[base + 8],  P[base + 9]  };
    BN bns{ P[base + 11], P[base + 10], P[base + 12], P[base + 13] };
    // conv1 -> h1 (bn + elu)
    conv_pass(xin, n, cin,  cout, 25, P[base + 18], P[base + 15], P[base + 0],  bn1,
              pseudo, src, dst, ev, cnt, yh, wpack, xr, agg, h1, 1, E, stream);
    // conv2 -> h2 (bn)
    conv_pass(h1,  n, cout, cout, 25, P[base + 19], P[base + 16], P[base + 1],  bn2,
              pseudo, src, dst, ev, cnt, yh, wpack, xr, agg, h2, 0, E, stream);
    // shortcut conv -> sb (bn)
    conv_pass(xin, n, cin,  cout, 1,  P[base + 20], P[base + 17], P[base + 14], bns,
              pseudo, src, dst, ev, cnt, yh, wpack, xr, agg, sb, 0, E, stream);
    int nC = n * cout;
    k_add_elu<<<(nC + 255) / 256, 256, 0, stream>>>(h2, sb, out, nC);
}

extern "C" void kernel_launch(void* const* d_in, const int* in_sizes, int n_in,
                              void* d_out, int out_size, void* d_ws, size_t ws_size,
                              hipStream_t stream) {
    const int N  = in_sizes[0];          // 65536 nodes
    const int E  = in_sizes[3] / 2;      // 524288 edges
    const int NC1 = 64 * 17 * 17;        // 18496
    const int NC2 = 64 * 9 * 9;          // 5184
    const int NP  = 64 * 25;             // 1600

    const float* x_in  = (const float*)d_in[0];
    const float* pos   = (const float*)d_in[1];
    const float* eattr = (const float*)d_in[2];
    const int*   eidx  = (const int*)d_in[3];
    const int*   batch0 = (const int*)d_in[4];
    const int* src0 = eidx;
    const int* dst0 = eidx + E;

    const float* P[80];
    for (int i = 0; i < n_in && i < 80; ++i) P[i] = (const float*)d_in[i];
    // pytree (alphabetical) flattening of params:
    // 5-8: bnf {b,g,m,v}; 9: fc1b; 10: fc1w; 11: fc2b; 12: fc2w;
    // 13: rb1 (21 leaves), 34: rb2, 55: rb3
    const int RB1 = 13, RB2 = 34, RB3 = 55;

    // -------- workspace carve (bump allocator, 256B aligned) --------
    size_t off = 0;
    char* base = (char*)d_ws;
    auto alloc = [&](size_t bytes) -> void* {
        void* p = base + off;
        off += (bytes + 255) & ~(size_t)255;
        return p;
    };
    _Float16* yh    = (_Float16*)alloc((size_t)N * 25 * 32 * 2);   // largest y table
    _Float16* wpack = (_Float16*)alloc((size_t)512 * 1024 * 2);    // packed weights (<=800KB)
    float* x0   = (float*)alloc((size_t)N * 32 * 4);               // level outputs (reused)
    float* h1   = (float*)alloc((size_t)N * 32 * 4);
    float* h2   = (float*)alloc((size_t)N * 32 * 4);
    float* sb   = (float*)alloc((size_t)N * 32 * 4);
    float* xr   = (float*)alloc((size_t)N * 32 * 4);
    float* agg  = (float*)alloc((size_t)N * 32 * 4);
    float* cnt  = (float*)alloc((size_t)N * 4);
    int*   clb  = (int*)alloc((size_t)N * 4);
    float* xc1   = (float*)alloc((size_t)NC1 * 32 * 4);
    float* posc1 = (float*)alloc((size_t)NC1 * 8);
    float* cnt1  = (float*)alloc((size_t)NC1 * 4);
    float* cva1  = (float*)alloc((size_t)NC1 * 4);
    int*   src1  = (int*)alloc((size_t)E * 4);
    int*   dst1  = (int*)alloc((size_t)E * 4);
    float* ev1   = (float*)alloc((size_t)E * 4);
    float* ps1   = (float*)alloc((size_t)E * 8);
    float* xc2   = (float*)alloc((size_t)NC2 * 64 * 4);
    float* posc2 = (float*)alloc((size_t)NC2 * 8);
    float* cnt2  = (float*)alloc((size_t)NC2 * 4);
    float* cva2  = (float*)alloc((size_t)NC2 * 4);
    int*   src2  = (int*)alloc((size_t)E * 4);
    int*   dst2  = (int*)alloc((size_t)E * 4);
    float* ev2   = (float*)alloc((size_t)E * 4);
    float* ps2   = (float*)alloc((size_t)E * 8);
    unsigned long long* hashT = (unsigned long long*)alloc((size_t)HASH_ENTRIES * 8);
    float* mx   = (float*)alloc(256);
    float* xp   = (float*)alloc((size_t)NP * 128 * 4);
    float* fco  = (float*)alloc((size_t)64 * 128 * 4);
    float* fca  = (float*)alloc((size_t)64 * 128 * 4);
    (void)ws_size; (void)out_size;

    const int TB = 256;

    // ================= res block 1 (level 0: N nodes, 1 -> 32) =================
    run_res_block(x_in, N, 1, 32, eattr, src0, dst0, /*ev=*/nullptr, P, RB1,
                  yh, wpack, xr, agg, cnt, h1, h2, sb, x0, E, stream);

    // ================= voxel pool 1 (size 2, 17x17) =================
    (void)hipMemsetAsync(cnt1, 0, (size_t)NC1 * 4, stream);
    (void)hipMemsetAsync(posc1, 0, (size_t)NC1 * 8, stream);
    k_fill<<<1024, TB, 0, stream>>>(xc1, -INFINITY, (long)NC1 * 32);
    k_pool_scatter<<<(N + TB - 1) / TB, TB, 0, stream>>>(x0, pos, nullptr, batch0, 0,
                                                         2.0f, 17, 32, xc1, posc1, cnt1, clb, N);
    k_pool_finish<<<(NC1 + TB - 1) / TB, TB, 0, stream>>>(xc1, posc1, cnt1, cva1, 32, NC1);
    (void)hipMemsetAsync(hashT, 0, (size_t)HASH_ENTRIES * 8, stream);
    (void)hipMemsetAsync(mx, 0, 16, stream);
    k_edge_remap<<<(E + TB - 1) / TB, TB, 0, stream>>>(src0, dst0, nullptr, clb, posc1, NC1,
                                                       src1, dst1, ev1, ps1, hashT, mx, E);
    k_pseudo_norm<<<(E + TB - 1) / TB, TB, 0, stream>>>(ps1, mx, E);

    // ================= res block 2 (NC1 nodes, 32 -> 64) =================
    run_res_block(xc1, NC1, 32, 64, ps1, src1, dst1, ev1, P, RB2,
                  yh, wpack, xr, agg, cnt, h1, h2, sb, x0, E, stream);

    // ================= voxel pool 2 (size 4, 9x9) =================
    (void)hipMemsetAsync(cnt2, 0, (size_t)NC2 * 4, stream);
    (void)hipMemsetAsync(posc2, 0, (size_t)NC2 * 8, stream);
    k_fill<<<1024, TB, 0, stream>>>(xc2, -INFINITY, (long)NC2 * 64);
    k_pool_scatter<<<(NC1 + TB - 1) / TB, TB, 0, stream>>>(x0, posc1, cva1, nullptr, 289,
                                                           4.0f, 9, 64, xc2, posc2, cnt2, clb, NC1);
    k_pool_finish<<<(NC2 + TB - 1) / TB, TB, 0, stream>>>(xc2, posc2, cnt2, cva2, 64, NC2);
    (void)hipMemsetAsync(hashT, 0, (size_t)HASH_ENTRIES * 8, stream);
    (void)hipMemsetAsync(mx, 0, 16, stream);
    k_edge_remap<<<(E + TB - 1) / TB, TB, 0, stream>>>(src1, dst1, ev1, clb, posc2, NC2,
                                                       src2, dst2, ev2, ps2, hashT, mx, E);
    k_pseudo_norm<<<(E + TB - 1) / TB, TB, 0, stream>>>(ps2, mx, E);

    // ================= res block 3 (NC2 nodes, 64 -> 128) =================
    run_res_block(xc2, NC2, 64, 128, ps2, src2, dst2, ev2, P, RB3,
                  yh, wpack, xr, agg, cnt, h1, h2, sb, x0, E, stream);

    // ================= final 5x5 max pool =================
    (void)hipMemsetAsync(cnt, 0, (size_t)NP * 4, stream);
    k_fill<<<1024, TB, 0, stream>>>(xp, -INFINITY, (long)NP * 128);
    k_pool_scatter<<<(NC2 + TB - 1) / TB, TB, 0, stream>>>(x0, posc2, cva2, nullptr, 81,
                                                           7.0f, 5, 128, xp, nullptr, cnt, clb, NC2);
    k_pool_finish<<<(NP + TB - 1) / TB, TB, 0, stream>>>(xp, nullptr, cnt, nullptr, 128, NP);

    // ================= FC1 (64x3200 @ 3200x128, WMMA) + ELU + BN =================
    launch_transform(xp, P[10], fco, false, 64, 3200, 128, 1, wpack, stream);
    {
        int n = 64 * 128;
        k_fc1_ep<<<(n + TB - 1) / TB, TB, 0, stream>>>(fco, P[9], P[6], P[5], P[7], P[8], fca, n);
    }

    // ================= FC2 + log_softmax =================
    k_head<<<1, 64, 0, stream>>>(fca, P[12], P[11], (float*)d_out);
}
